// MiniTransformer_25709674234175
// MI455X (gfx1250) — matrix operand order repaired and checked by reading, not compiled, not run on hardware
//
#include <hip/hip_runtime.h>
#include <math.h>

// ---------------------------------------------------------------------------
// MiniTransformer forward for gfx1250 (MI455X): bf16 WMMA + TDM/async staging.
// L=4, B=2, T=2048, D=1024, H=16, HD=64, F=4096, V=32000.
// ---------------------------------------------------------------------------

#define L_ 4
#define B_ 2
#define T_ 2048
#define D_ 1024
#define H_ 16
#define HD_ 64
#define F_ 4096
#define V_ 32000
#define M_ (B_ * T_)          // 4096 rows of activations
#define EPS_ 1e-5f

typedef __bf16 bf16_t;
typedef bf16_t v16bf __attribute__((ext_vector_type(16)));
typedef bf16_t v8bf  __attribute__((ext_vector_type(8)));
typedef float  v8f   __attribute__((ext_vector_type(8)));
typedef unsigned int u32x4 __attribute__((ext_vector_type(4)));
typedef int i32x4 __attribute__((ext_vector_type(4)));
typedef int i32x8 __attribute__((ext_vector_type(8)));

#define HAVE_TDM __has_builtin(__builtin_amdgcn_tensor_load_to_lds)

static __device__ __forceinline__ v16bf cat16(v8bf lo, v8bf hi) {
  v16bf r;
#pragma unroll
  for (int i = 0; i < 8; ++i) { r[i] = lo[i]; r[i + 8] = hi[i]; }
  return r;
}

static __device__ __forceinline__ v8f zero8() {
  v8f z;
#pragma unroll
  for (int i = 0; i < 8; ++i) z[i] = 0.0f;
  return z;
}

static __device__ __forceinline__ v8f wmma_bf16(v16bf a, v16bf b, v8f c) {
  // D = A(16x32 bf16) x B(32x16 bf16) + C(16x16 f32)
  return __builtin_amdgcn_wmma_f32_16x16x32_bf16(
      /*neg_a=*/false, a, /*neg_b=*/false, b,
      /*c_mod=*/(short)0, c, /*reuse_a=*/false, /*reuse_b=*/false);
}

// Async 16B global->LDS copy (ASYNCcnt); lds_off = LDS byte offset (addr[31:0]).
static __device__ __forceinline__ void async_b128(unsigned lds_off, const void* g) {
  asm volatile("global_load_async_to_lds_b128 %0, %1, off"
               :: "v"(lds_off), "v"(g) : "memory");
}
static __device__ __forceinline__ void wait_async0() {
  asm volatile("s_wait_asynccnt 0x0" ::: "memory");
}

// ---------------------------------------------------------------------------
// fp32 -> bf16 conversion (weights)
// ---------------------------------------------------------------------------
__global__ __launch_bounds__(256) void cvt_bf16_kernel(
    const float* __restrict__ in, bf16_t* __restrict__ out, long n) {
  long i = (long)blockIdx.x * 256 + threadIdx.x;
  if (i < n) out[i] = (bf16_t)in[i];
}

// ---------------------------------------------------------------------------
// x[b,t,:] = tok_emb[tokens[b,t]] + pos_emb[t]
// ---------------------------------------------------------------------------
__global__ __launch_bounds__(256) void embed_kernel(
    const int* __restrict__ tokens, const float* __restrict__ tok_emb,
    const float* __restrict__ pos_emb, float* __restrict__ x) {
  const int row = blockIdx.x;          // b*T + t
  const int t = row & (T_ - 1);
  const int tok = tokens[row];
  const int c = threadIdx.x * 4;
  float4 a = *(const float4*)(tok_emb + (long)tok * D_ + c);
  float4 p = *(const float4*)(pos_emb + (long)t * D_ + c);
  float4 o; o.x = a.x + p.x; o.y = a.y + p.y; o.z = a.z + p.z; o.w = a.w + p.w;
  *(float4*)(x + (long)row * D_ + c) = o;
}

// ---------------------------------------------------------------------------
// LayerNorm over D=1024, one block (256 thr) per row, bf16 output.
// ---------------------------------------------------------------------------
__global__ __launch_bounds__(256) void layernorm_kernel(
    const float* __restrict__ x, const float* __restrict__ g,
    const float* __restrict__ b, bf16_t* __restrict__ out) {
  __shared__ float red[8];
  const int row = blockIdx.x;
  const int tid = threadIdx.x;
  const int lane = tid & 31, wave = tid >> 5;
  const float4 v = *(const float4*)(x + (long)row * D_ + tid * 4);

  float s = v.x + v.y + v.z + v.w;
#pragma unroll
  for (int off = 1; off < 32; off <<= 1) s += __shfl_xor(s, off, 32);
  if (lane == 0) red[wave] = s;
  __syncthreads();
  s = 0.0f;
#pragma unroll
  for (int i = 0; i < 8; ++i) s += red[i];
  const float mean = s * (1.0f / D_);
  __syncthreads();

  float dx[4] = {v.x - mean, v.y - mean, v.z - mean, v.w - mean};
  float q = dx[0]*dx[0] + dx[1]*dx[1] + dx[2]*dx[2] + dx[3]*dx[3];
#pragma unroll
  for (int off = 1; off < 32; off <<= 1) q += __shfl_xor(q, off, 32);
  if (lane == 0) red[wave] = q;
  __syncthreads();
  q = 0.0f;
#pragma unroll
  for (int i = 0; i < 8; ++i) q += red[i];
  const float rstd = rsqrtf(q * (1.0f / D_) + EPS_);

#pragma unroll
  for (int i = 0; i < 4; ++i) {
    const int col = tid * 4 + i;
    out[(long)row * D_ + col] = (bf16_t)(dx[i] * rstd * g[col] + b[col]);
  }
}

// ---------------------------------------------------------------------------
// bf16 GEMM: Out[M,N] = epilogue( A[M,K] @ W[K,N] ).
// Block tile 64x128, 8 waves, each wave 32x32 (2x2 WMMA accumulators).
// A tile (64x32, block-uniform) staged via the Tensor Data Mover when the
// builtin exists (one descriptor per k-step, issued by wave 0, TENSORcnt),
// else via global_load_async_to_lds_b128. B tile (32x128) staged row-major
// with async b128 copies (ASYNCcnt). A fragments read as aligned
// ds_load_b128 pairs in the documented 16-bit A-layout; B fragments read
// per-lane from the [k][n] tile.
// Assumes M%64==0, N%128==0, K%32==0 (true for all shapes here).
// ---------------------------------------------------------------------------
template <bool BIAS, bool GELU, bool RES, bool OUT_BF16>
__global__ __launch_bounds__(256) void gemm_bf16_kernel(
    const bf16_t* __restrict__ A, const bf16_t* __restrict__ W,
    const float* __restrict__ bias, const float* __restrict__ Cres,
    void* __restrict__ Out, int M, int N, int K) {
  __shared__ __align__(16) bf16_t sA[64 * 32];    // [m][k]
  __shared__ __align__(16) bf16_t sB[32 * 128];   // [k][n]

  const int tid = threadIdx.x;
  const int lane = tid & 31, wave = tid >> 5;
  const int ln16 = lane & 15, hi = lane >> 4;
  const int wm = wave >> 2;            // 0..1 -> 32-row strip
  const int wn = wave & 3;             // 0..3 -> 32-col strip
  const long bm = (long)blockIdx.y * 64;
  const long bn = (long)blockIdx.x * 128;
  const unsigned sA_base = (unsigned)(uintptr_t)&sA[0];  // LDS byte offset
  const unsigned sB_base = (unsigned)(uintptr_t)&sB[0];

  v8f acc[2][2];
#pragma unroll
  for (int i = 0; i < 2; ++i)
#pragma unroll
    for (int j = 0; j < 2; ++j) acc[i][j] = zero8();

  for (int k0 = 0; k0 < K; k0 += 32) {
    // ---- stage A tile 64x32 ----
#if HAVE_TDM
    if (wave == 0) {
      // Tensor DMA descriptor: 2D tile 32(k) x 64(m), 2-byte elements.
      const unsigned long long ga =
          (unsigned long long)(uintptr_t)(A + bm * (long)K + k0);
      u32x4 g0; i32x8 g1; i32x4 g2, g3;
      g0[0] = 1u;                                   // count=1 valid descriptor
      g0[1] = sA_base;                              // lds_addr
      g0[2] = (unsigned)ga;                         // global_addr[31:0]
      g0[3] = (unsigned)((ga >> 32) & 0x01FFFFFFu)  // global_addr[56:32]
              | (2u << 30);                         // type=2 ("image")
      g1[0] = (int)(1u << 16);                      // data_size=1 (2 bytes)
      g1[1] = (int)(((unsigned)K & 0xFFFFu) << 16); // tensor_dim0[15:0]
      g1[2] = (int)((((unsigned)K >> 16) & 0xFFFFu) // tensor_dim0[31:16]
              | (((unsigned)M & 0xFFFFu) << 16));   // tensor_dim1[15:0]
      g1[3] = (int)((((unsigned)M >> 16) & 0xFFFFu) // tensor_dim1[31:16]
              | (32u << 16));                       // tile_dim0 = 32
      g1[4] = 64;                                   // tile_dim1=64, tile_dim2=0
      g1[5] = K;                                    // tensor_dim0_stride[31:0]
      g1[6] = 0;                                    // stride hi, dim1_stride lo
      g1[7] = 0;
#pragma unroll
      for (int i = 0; i < 4; ++i) { g2[i] = 0; g3[i] = 0; }
#if defined(__clang_major__) && __clang_major__ >= 23
      i32x8 g4;
#pragma unroll
      for (int i = 0; i < 8; ++i) g4[i] = 0;
      __builtin_amdgcn_tensor_load_to_lds(g0, g1, g2, g3, g4, 0);
#else
      __builtin_amdgcn_tensor_load_to_lds(g0, g1, g2, g3, 0);
#endif
      __builtin_amdgcn_s_wait_tensorcnt(0);
    }
#else
    {
      const int r = tid >> 2;               // 0..63
      const int c = (tid & 3) * 8;          // 0,8,16,24
      async_b128(sA_base + (unsigned)(r * 32 + c) * 2,
                 A + (bm + r) * (long)K + k0 + c);
    }
#endif
    // ---- stage B tile 32x128 row-major via async b128 (16 elems/thread) ----
    {
      const int kr = tid >> 3;              // 0..31
      const int nc = (tid & 7) * 16;        // 0,16,...,112
      const bf16_t* src = W + (long)(k0 + kr) * N + bn + nc;
      async_b128(sB_base + (unsigned)(kr * 128 + nc) * 2, src);
      async_b128(sB_base + (unsigned)(kr * 128 + nc + 8) * 2, src + 8);
      if (k0 + 32 < K)                      // prefetch next W tile into L2
        __builtin_prefetch(src + 32 * (long)N, 0, 1);
    }
    wait_async0();
    __syncthreads();

    // ---- fragments + 4 WMMAs ----
    v16bf af[2], bfr[2];
#pragma unroll
    for (int mi = 0; mi < 2; ++mi) {
      const int m = wm * 32 + mi * 16 + ln16;
      af[mi] = cat16(*(const v8bf*)(sA + m * 32 + hi * 8),
                     *(const v8bf*)(sA + m * 32 + 16 + hi * 8));
    }
#pragma unroll
    for (int ni = 0; ni < 2; ++ni) {
      const int n = wn * 32 + ni * 16 + ln16;
      v16bf t;
#pragma unroll
      for (int e = 0; e < 16; ++e) t[e] = sB[(hi * 8 + e + (e >= 8 ? 8 : 0)) * 128 + n];
      bfr[ni] = t;
    }
#pragma unroll
    for (int mi = 0; mi < 2; ++mi)
#pragma unroll
      for (int ni = 0; ni < 2; ++ni)
        acc[mi][ni] = wmma_bf16(af[mi], bfr[ni], acc[mi][ni]);
    __syncthreads();
  }

  // ---- epilogue: C layout = lane col ln16, rows r+8*hi ----
#pragma unroll
  for (int mi = 0; mi < 2; ++mi) {
#pragma unroll
    for (int ni = 0; ni < 2; ++ni) {
      const long ncol = bn + wn * 32 + ni * 16 + ln16;
      const float bv = BIAS ? bias[ncol] : 0.0f;
#pragma unroll
      for (int r = 0; r < 8; ++r) {
        const long mrow = bm + wm * 32 + mi * 16 + r + 8 * hi;
        float v = acc[mi][ni][r] + bv;
        if (GELU) v = 0.5f * v * (1.0f + erff(v * 0.70710678118654752f));
        if (RES) v += Cres[mrow * (long)N + ncol];
        if (OUT_BF16) ((bf16_t*)Out)[mrow * (long)N + ncol] = (bf16_t)v;
        else          ((float*)Out)[mrow * (long)N + ncol] = v;
      }
    }
  }
}

// ---------------------------------------------------------------------------
// Flash attention: qkv bf16 [B,T,3,H,HD] -> out bf16 [B,T,D].
// One wave per 16-query tile, online softmax, all matmuls via WMMA.
// K/V (16 MB bf16 per layer) stay resident in the 192 MB L2.
// ---------------------------------------------------------------------------
__global__ __launch_bounds__(256) void flash_attn_kernel(
    const bf16_t* __restrict__ qkv, bf16_t* __restrict__ out) {
  __shared__ __align__(16) bf16_t sP[8][16 * 32];  // per-wave P staging

  const int tid = threadIdx.x;
  const int lane = tid & 31, wave = tid >> 5;
  const int ln16 = lane & 15, hi = lane >> 4;
  const int bh = blockIdx.x;            // 0..B*H-1
  const int b = bh >> 4, h = bh & 15;
  const int q0 = blockIdx.y * 128 + wave * 16;
  const long rs = 3 * D_;               // qkv row stride (elements)

  const bf16_t* qbase = qkv + (long)b * T_ * rs + h * HD_;
  const bf16_t* kbase = qbase + D_;
  const bf16_t* vbase = qbase + 2 * D_;

  // Q A-fragments for the two 32-wide d-chunks of HD=64
  v16bf aq[2];
  {
    const bf16_t* qp = qbase + (long)(q0 + ln16) * rs;
#pragma unroll
    for (int c = 0; c < 2; ++c)
      aq[c] = cat16(*(const v8bf*)(qp + c * 32 + hi * 16),
                    *(const v8bf*)(qp + c * 32 + hi * 16 + 8));
  }

  v8f o[4];
#pragma unroll
  for (int i = 0; i < 4; ++i) o[i] = zero8();
  float mrow[8], lrow[8];
#pragma unroll
  for (int r = 0; r < 8; ++r) { mrow[r] = -3.0e38f; lrow[r] = 0.0f; }

  bf16_t* P = &sP[wave][0];

  for (int k0 = 0; k0 <= q0 + 15; k0 += 32) {
    // ---- S = (Q K^T) * scale, causal mask; two 16x16 f32 frags ----
    v8f s[2];
#pragma unroll
    for (int kt = 0; kt < 2; ++kt) {
      const int key = k0 + kt * 16 + ln16;
      v8f sc = zero8();
#pragma unroll
      for (int c = 0; c < 2; ++c) {
        const bf16_t* kp = kbase + (long)key * rs + c * 32 + hi * 16;
        sc = wmma_bf16(aq[c], cat16(*(const v8bf*)kp, *(const v8bf*)(kp + 8)), sc);
      }
#pragma unroll
      for (int r = 0; r < 8; ++r) {
        const int qrow = q0 + r + 8 * hi;
        sc[r] = (key <= qrow) ? sc[r] * 0.125f : -3.0e38f;  // 1/sqrt(64)
      }
      s[kt] = sc;
    }

    // ---- online softmax row statistics (cross-lane over 16 columns) ----
    float alpha[8];
#pragma unroll
    for (int r = 0; r < 8; ++r) {
      float mx = fmaxf(s[0][r], s[1][r]);
#pragma unroll
      for (int off = 1; off < 16; off <<= 1) mx = fmaxf(mx, __shfl_xor(mx, off, 32));
      const float mnew = fmaxf(mrow[r], mx);
      alpha[r] = __expf(mrow[r] - mnew);
      mrow[r] = mnew;
    }
#pragma unroll
    for (int kt = 0; kt < 2; ++kt)
#pragma unroll
      for (int r = 0; r < 8; ++r) s[kt][r] = __expf(s[kt][r] - mrow[r]);
#pragma unroll
    for (int r = 0; r < 8; ++r) {
      float ps = s[0][r] + s[1][r];
#pragma unroll
      for (int off = 1; off < 16; off <<= 1) ps += __shfl_xor(ps, off, 32);
      lrow[r] = lrow[r] * alpha[r] + ps;
    }
#pragma unroll
    for (int nc = 0; nc < 4; ++nc)
#pragma unroll
      for (int r = 0; r < 8; ++r) o[nc][r] *= alpha[r];

    // ---- C-layout -> A-layout conversion of P through wave-local LDS ----
#pragma unroll
    for (int kt = 0; kt < 2; ++kt)
#pragma unroll
      for (int r = 0; r < 8; ++r)
        P[(r + 8 * hi) * 32 + kt * 16 + ln16] = (bf16_t)s[kt][r];
    asm volatile("s_wait_dscnt 0x0" ::: "memory");  // wave-local DS ordering
    const bf16_t* pp = P + ln16 * 32;
    const v16bf ap = cat16(*(const v8bf*)(pp + hi * 16),
                           *(const v8bf*)(pp + hi * 16 + 8));

    // ---- O += P @ V (four 16-col chunks of HD) ----
#pragma unroll
    for (int nc = 0; nc < 4; ++nc) {
      const int d = nc * 16 + ln16;
      v16bf bv;
#pragma unroll
      for (int e = 0; e < 16; ++e)
        bv[e] = vbase[(long)(k0 + hi * 16 + e) * rs + d];
      o[nc] = wmma_bf16(ap, bv, o[nc]);
    }
  }

  // ---- normalize and write O as bf16 [B,T,D] ----
#pragma unroll
  for (int nc = 0; nc < 4; ++nc)
#pragma unroll
    for (int r = 0; r < 8; ++r) {
      const int qrow = q0 + r + 8 * hi;
      const float v = o[nc][r] / lrow[r];
      out[((long)b * T_ + qrow) * D_ + h * HD_ + nc * 16 + ln16] = (bf16_t)v;
    }
}

// ---------------------------------------------------------------------------
// Host orchestration
// ---------------------------------------------------------------------------
extern "C" void kernel_launch(void* const* d_in, const int* in_sizes, int n_in,
                              void* d_out, int out_size, void* d_ws, size_t ws_size,
                              hipStream_t stream) {
  const int*   tokens  = (const int*)  d_in[0];
  const float* tok_emb = (const float*)d_in[1];
  const float* pos_emb = (const float*)d_in[2];
  const float* qkv_w   = (const float*)d_in[3];
  const float* out_w   = (const float*)d_in[4];
  const float* ln1_g   = (const float*)d_in[5];
  const float* ln1_b   = (const float*)d_in[6];
  const float* ln2_g   = (const float*)d_in[7];
  const float* ln2_b   = (const float*)d_in[8];
  const float* ff_w1   = (const float*)d_in[9];
  const float* ff_b1   = (const float*)d_in[10];
  const float* ff_w2   = (const float*)d_in[11];
  const float* ff_b2   = (const float*)d_in[12];
  const float* lnf_g   = (const float*)d_in[13];
  const float* lnf_b   = (const float*)d_in[14];
  const float* head_w  = (const float*)d_in[15];
  float* logits = (float*)d_out;

  // ---- workspace carve-up (256B aligned) ----
  char* w = (char*)d_ws;
  auto carve = [&](size_t bytes) { char* p = w; w += (bytes + 255) & ~(size_t)255; return p; };
  bf16_t* wqkv = (bf16_t*)carve((size_t)L_ * D_ * 3 * D_ * 2);
  bf16_t* wout = (bf16_t*)carve((size_t)L_ * D_ * D_ * 2);
  bf16_t* wff1 = (bf16_t*)carve((size_t)L_ * D_ * F_ * 2);
  bf16_t* wff2 = (bf16_t*)carve((size_t)L_ * F_ * D_ * 2);
  bf16_t* whead= (bf16_t*)carve((size_t)D_ * V_ * 2);
  float*  x    = (float*) carve((size_t)M_ * D_ * 4);
  bf16_t* hb   = (bf16_t*)carve((size_t)M_ * D_ * 2);
  bf16_t* qkvb = (bf16_t*)carve((size_t)M_ * 3 * D_ * 2);
  bf16_t* atto = (bf16_t*)carve((size_t)M_ * D_ * 2);
  bf16_t* ffh  = (bf16_t*)carve((size_t)M_ * F_ * 2);

  auto cvt = [&](const float* src, bf16_t* dst, long n) {
    cvt_bf16_kernel<<<dim3((unsigned)((n + 255) / 256)), 256, 0, stream>>>(src, dst, n);
  };
  cvt(qkv_w, wqkv, (long)L_ * D_ * 3 * D_);
  cvt(out_w, wout, (long)L_ * D_ * D_);
  cvt(ff_w1, wff1, (long)L_ * D_ * F_);
  cvt(ff_w2, wff2, (long)L_ * F_ * D_);
  cvt(head_w, whead, (long)D_ * V_);

  embed_kernel<<<dim3(M_), 256, 0, stream>>>(tokens, tok_emb, pos_emb, x);

  for (int l = 0; l < L_; ++l) {
    const bf16_t* Wqkv = wqkv + (long)l * D_ * 3 * D_;
    const bf16_t* Wout = wout + (long)l * D_ * D_;
    const bf16_t* Wf1  = wff1 + (long)l * D_ * F_;
    const bf16_t* Wf2  = wff2 + (long)l * F_ * D_;

    // h = LN1(x)  (bf16)
    layernorm_kernel<<<dim3(M_), 256, 0, stream>>>(x, ln1_g + l * D_, ln1_b + l * D_, hb);
    // qkv = h @ Wqkv  (bf16 out)
    gemm_bf16_kernel<false,false,false,true><<<dim3(3*D_/128, M_/64), 256, 0, stream>>>(
        hb, Wqkv, nullptr, nullptr, qkvb, M_, 3*D_, D_);
    // attention (flash, causal)
    flash_attn_kernel<<<dim3(B_*H_, T_/128), 256, 0, stream>>>(qkvb, atto);
    // x = x + attn_out @ Wout  (f32 out, residual)
    gemm_bf16_kernel<false,false,true,false><<<dim3(D_/128, M_/64), 256, 0, stream>>>(
        atto, Wout, nullptr, x, x, M_, D_, D_);
    // h2 = LN2(x)
    layernorm_kernel<<<dim3(M_), 256, 0, stream>>>(x, ln2_g + l * D_, ln2_b + l * D_, hb);
    // ffh = gelu(h2 @ Wf1 + b1)  (bf16 out)
    gemm_bf16_kernel<true,true,false,true><<<dim3(F_/128, M_/64), 256, 0, stream>>>(
        hb, Wf1, ff_b1 + l * F_, nullptr, ffh, M_, F_, D_);
    // x = x + ffh @ Wf2 + b2  (f32 out, residual)
    gemm_bf16_kernel<true,false,true,false><<<dim3(D_/128, M_/64), 256, 0, stream>>>(
        ffh, Wf2, ff_b2 + l * D_, x, x, M_, D_, F_);
  }

  // final LN + head
  layernorm_kernel<<<dim3(M_), 256, 0, stream>>>(x, lnf_g, lnf_b, hb);
  gemm_bf16_kernel<false,false,false,false><<<dim3(V_/128, M_/64), 256, 0, stream>>>(
      hb, whead, nullptr, nullptr, logits, M_, V_, D_);
}
